// MoEBlock_73048803770960
// MI455X (gfx1250) — matrix operand order repaired and checked by reading, not compiled, not run on hardware
//
#include <hip/hip_runtime.h>
#include <hip/hip_bf16.h>

// MoE block: B=1024 tokens, M=10000 model dim, E=8 experts, H=512 hidden, top-2.
// Sparse routing + bf16x3 WMMA GEMMs (v_wmma_f32_16x16x32_bf16), fp32 accumulate.

#define B_TOK 1024
#define M_DIM 10000
#define E_EXP 8
#define H_DIM 512

typedef __attribute__((ext_vector_type(16))) __bf16 v16bf;
typedef __attribute__((ext_vector_type(8)))  float  v8f;

// Split two fp32 into (hi,lo) bf16 pairs and store as two packed dwords.
__device__ __forceinline__ void stagePair(void* pHi, void* pLo, float f0, float f1) {
  __bf16 h0 = (__bf16)f0, h1 = (__bf16)f1;
  __bf16 l0 = (__bf16)(f0 - (float)h0), l1 = (__bf16)(f1 - (float)h1);
  union { __bf16 b[2]; unsigned u; } uh, ul;
  uh.b[0] = h0; uh.b[1] = h1;
  ul.b[0] = l0; ul.b[1] = l1;
  *(unsigned*)pHi = uh.u;
  *(unsigned*)pLo = ul.u;
}

// ---------------------------------------------------------------------------
// Router: logits[b,e] = x[b,:] . Wr[e,:]; top-2 softmax; scatter into
// per-expert (token, weight) lists.
// ---------------------------------------------------------------------------
__global__ __launch_bounds__(256) void router_kernel(
    const float* __restrict__ x, const float* __restrict__ Wr,
    int* __restrict__ counts, int* __restrict__ perm, float* __restrict__ pw) {
  const int b = blockIdx.x;
  const int t = threadIdx.x;
  float acc[E_EXP];
#pragma unroll
  for (int e = 0; e < E_EXP; ++e) acc[e] = 0.f;
  for (int k = t; k < M_DIM; k += 256) {
    float xv = x[b * M_DIM + k];
#pragma unroll
    for (int e = 0; e < E_EXP; ++e) acc[e] += xv * Wr[e * M_DIM + k];
  }
  __shared__ float red[E_EXP][256];
#pragma unroll
  for (int e = 0; e < E_EXP; ++e) red[e][t] = acc[e];
  __syncthreads();
  for (int off = 128; off > 0; off >>= 1) {
    if (t < off) {
#pragma unroll
      for (int e = 0; e < E_EXP; ++e) red[e][t] += red[e][t + off];
    }
    __syncthreads();
  }
  if (t == 0) {
    float lg[E_EXP];
#pragma unroll
    for (int e = 0; e < E_EXP; ++e) lg[e] = red[e][0];
    int i0 = 0;
    for (int e = 1; e < E_EXP; ++e) if (lg[e] > lg[i0]) i0 = e;
    int i1 = -1;
    for (int e = 0; e < E_EXP; ++e) {
      if (e == i0) continue;
      if (i1 < 0 || lg[e] > lg[i1]) i1 = e;
    }
    float mx = lg[i0];
    float e0 = __expf(lg[i0] - mx), e1 = __expf(lg[i1] - mx);
    float inv = 1.f / (e0 + e1);
    int p0 = atomicAdd(&counts[i0], 1);
    perm[i0 * B_TOK + p0] = b;  pw[i0 * B_TOK + p0] = e0 * inv;
    int p1 = atomicAdd(&counts[i1], 1);
    perm[i1 * B_TOK + p1] = b;  pw[i1 * B_TOK + p1] = e1 * inv;
  }
}

// ---------------------------------------------------------------------------
// GEMM1: h[e][slot][:] = relu(x[perm[e][slot],:] @ W1[e]^T + b1[e])
// Block = 256 threads (8 waves). Tile = 64 slots x 128 H cols. Kc = 32.
// Wave w: row-tile rt = w/2, col-tiles [4*(w&1) .. +4). 3 WMMAs/tile (bf16x3).
// ---------------------------------------------------------------------------
__global__ __launch_bounds__(256) void gemm1_kernel(
    const float* __restrict__ x, const float* __restrict__ W1,
    const float* __restrict__ b1, const int* __restrict__ counts,
    const int* __restrict__ perm, float* __restrict__ hbuf) {
  const int e = blockIdx.z;
  const int n_e = counts[e];
  const int slot0 = blockIdx.x * 64;
  if (slot0 >= n_e) return;
  const int hb = blockIdx.y * 128;
  const int t = threadIdx.x;
  const int w = t >> 5, l = t & 31;

  __shared__ int   sperm[64];
  __shared__ v16bf aBuf[2][4][32];   // [hi|lo][row-tile][lane], WMMA A lane order
  __shared__ v16bf bBuf[2][8][32];   // [hi|lo][col-tile][lane], WMMA B lane order

  if (t < 64) {
    int s = slot0 + t;
    sperm[t] = (s < n_e) ? perm[e * B_TOK + s] : -1;
  }
  __syncthreads();

  // ---- Hoisted staging assignments (fixed per thread across the K loop) ----
  // A pairs: i = t + 256*j (j<4): s = i>>4 (slot-in-tile), kp = i&15 (k-pair).
  unsigned aOff[4], aLds[4];
  bool aVal[4], aTl[4];
#pragma unroll
  for (int j = 0; j < 4; ++j) {
    int i = t + 256 * j;
    int s = i >> 4, kp = i & 15;
    int row = sperm[s];
    aVal[j] = (row >= 0);
    aTl[j]  = (row >= 0) && (kp < 8);            // tail slice: k0=9984, kk<10000
    unsigned r = (row < 0) ? 0u : (unsigned)row;
    aOff[j] = (r * (unsigned)M_DIM + 2u * (unsigned)kp) * 4u;
    // A 16x32 bf16 lane order: lanes 0-15 hold K 0..7 (idx0-7) & 16..23 (idx8-15)
    int lane = (s & 15) + (((kp & 7) < 4) ? 0 : 16);
    int idx  = ((2 * kp) & 7) + ((kp >= 8) ? 8 : 0);
    aLds[j] = (unsigned)(((((s >> 4) * 32) + lane) * 16 + idx) * 2);
  }
  // B pairs: i = t + 256*j (j<8): c = i>>4 (col), kp = i&15.
  unsigned bOff[8], bLds[8];
  bool bTl[8];
#pragma unroll
  for (int j = 0; j < 8; ++j) {
    int i = t + 256 * j;
    int c = i >> 4, kp = i & 15;
    bTl[j] = (kp < 8);
    bOff[j] = ((unsigned)(e * H_DIM + hb + c) * (unsigned)M_DIM + 2u * (unsigned)kp) * 4u;
    // B 32x16 bf16 lane order: lanes 0-15 hold K 0..7 (idx0-7) & 16..23 (idx8-15)
    int lane = (c & 15) + (((kp & 7) < 4) ? 0 : 16);
    int idx  = ((2 * kp) & 7) + ((kp >= 8) ? 8 : 0);
    bLds[j] = (unsigned)(((((c >> 4) * 32) + lane) * 16 + idx) * 2);
  }

  const char* xB  = (const char*)x;
  const char* w1B = (const char*)W1;
  char* aHiB = (char*)&aBuf[0][0][0];
  char* aLoB = (char*)&aBuf[1][0][0];
  char* bHiB = (char*)&bBuf[0][0][0];
  char* bLoB = (char*)&bBuf[1][0][0];

  const int rt = w >> 1;
  const int ct0 = (w & 1) * 4;
  v8f acc[4] = {};

  auto consume = [&]() {
    v16bf ah = aBuf[0][rt][l];
    v16bf al = aBuf[1][rt][l];
#pragma unroll
    for (int c = 0; c < 4; ++c) {
      v16bf bh = bBuf[0][ct0 + c][l];
      v16bf bl = bBuf[1][ct0 + c][l];
      acc[c] = __builtin_amdgcn_wmma_f32_16x16x32_bf16(false, ah, false, bh, (short)0, acc[c], false, false);
      acc[c] = __builtin_amdgcn_wmma_f32_16x16x32_bf16(false, al, false, bh, (short)0, acc[c], false, false);
      acc[c] = __builtin_amdgcn_wmma_f32_16x16x32_bf16(false, ah, false, bl, (short)0, acc[c], false, false);
    }
  };

  // ---- Hot loop: 312 full K-slices, no bounds logic ----
  for (int ks = 0; ks < (M_DIM / 32); ++ks) {
    const unsigned kb = (unsigned)ks * 128u;
    __syncthreads();
#pragma unroll
    for (int j = 0; j < 4; ++j) {
      float2 f = *(const float2*)(xB + (aOff[j] + kb));
      float f0 = aVal[j] ? f.x : 0.f, f1 = aVal[j] ? f.y : 0.f;
      stagePair(aHiB + aLds[j], aLoB + aLds[j], f0, f1);
    }
#pragma unroll
    for (int j = 0; j < 8; ++j) {
      float2 f = *(const float2*)(w1B + (bOff[j] + kb));
      stagePair(bHiB + bLds[j], bLoB + bLds[j], f.x, f.y);
    }
    __syncthreads();
    consume();
  }
  // ---- Peeled tail slice (k0 = 9984, only kp<8 in-range; clamp addresses) ----
  {
    const unsigned kb = (unsigned)(M_DIM / 32) * 128u;
    __syncthreads();
#pragma unroll
    for (int j = 0; j < 4; ++j) {
      unsigned off = aTl[j] ? (aOff[j] + kb) : 0u;
      float2 f = *(const float2*)(xB + off);
      float f0 = aTl[j] ? f.x : 0.f, f1 = aTl[j] ? f.y : 0.f;
      stagePair(aHiB + aLds[j], aLoB + aLds[j], f0, f1);
    }
#pragma unroll
    for (int j = 0; j < 8; ++j) {
      unsigned off = bTl[j] ? (bOff[j] + kb) : 0u;
      float2 f = *(const float2*)(w1B + off);
      float f0 = bTl[j] ? f.x : 0.f, f1 = bTl[j] ? f.y : 0.f;
      stagePair(bHiB + bLds[j], bLoB + bLds[j], f0, f1);
    }
    __syncthreads();
    consume();
  }

  // ---- Epilogue: bias + relu -> hbuf. lane l -> col l%16; vgpr r -> row r+8*(l/16). ----
  const int n = l & 15;
  const int rbase = (l >> 4) * 8;
#pragma unroll
  for (int c = 0; c < 4; ++c) {
    int col = hb + (ct0 + c) * 16 + n;
    float bias = b1[e * H_DIM + col];
#pragma unroll
    for (int r = 0; r < 8; ++r) {
      int slot = slot0 + rt * 16 + rbase + r;
      if (slot < n_e) {
        float v = acc[c][r] + bias;
        hbuf[(size_t)(e * B_TOK + slot) * H_DIM + col] = v > 0.f ? v : 0.f;
      }
    }
  }
}

// ---------------------------------------------------------------------------
// GEMM2: out[tok,:] += w * (h[e][slot,:] @ W2[e]^T + b2[e])
// Block = 160 threads (5 waves). Tile = 64 slots x 80 M cols (10000 = 125*80).
// Wave w owns col-tile w and all 4 row-tiles. K = 512 = 16 slices of 32.
// A staged by threads 0-127 (8 pairs each); B staged by all 160 (8 pairs each).
// ---------------------------------------------------------------------------
__global__ __launch_bounds__(160) void gemm2_kernel(
    const float* __restrict__ hbuf, const float* __restrict__ W2,
    const float* __restrict__ b2, const int* __restrict__ counts,
    const int* __restrict__ perm, const float* __restrict__ pw,
    float* __restrict__ out) {
  const int e = blockIdx.z;
  const int n_e = counts[e];
  const int slot0 = blockIdx.x * 64;
  if (slot0 >= n_e) return;
  const int mb = blockIdx.y * 80;
  const int t = threadIdx.x;
  const int w = t >> 5, l = t & 31;

  __shared__ int   stok[64];
  __shared__ float swgt[64];
  __shared__ v16bf aBuf[2][4][32];
  __shared__ v16bf bBuf[2][5][32];

  if (t < 64) {
    int s = slot0 + t;
    bool v = (s < n_e);
    stok[t] = v ? perm[e * B_TOK + s] : -1;
    swgt[t] = v ? pw[e * B_TOK + s] : 0.f;
  }

  // A pairs: i = t + 128*j (j<8, threads 0-127): s = i>>4, kp = i&15. 1024 items.
  unsigned aOff[8], aLds[8];
  bool aVal[8];
  if (t < 128) {
#pragma unroll
    for (int j = 0; j < 8; ++j) {
      int i = t + 128 * j;
      int s = i >> 4, kp = i & 15;
      int slot = slot0 + s;
      aVal[j] = (slot < n_e);
      unsigned rs = aVal[j] ? (unsigned)(e * B_TOK + slot) : 0u;
      aOff[j] = (rs * (unsigned)H_DIM + 2u * (unsigned)kp) * 4u;
      int lane = (s & 15) + (((kp & 7) < 4) ? 0 : 16);
      int idx  = ((2 * kp) & 7) + ((kp >= 8) ? 8 : 0);
      aLds[j] = (unsigned)(((((s >> 4) * 32) + lane) * 16 + idx) * 2);
    }
  }
  // B pairs: i = t + 160*j (j<8): c = i>>4, kp = i&15. 1280 items (80 cols).
  unsigned bOff[8], bLds[8];
#pragma unroll
  for (int j = 0; j < 8; ++j) {
    int i = t + 160 * j;
    int c = i >> 4, kp = i & 15;
    bOff[j] = ((unsigned)(e * M_DIM + mb + c) * (unsigned)H_DIM + 2u * (unsigned)kp) * 4u;
    int lane = (c & 15) + (((kp & 7) < 4) ? 0 : 16);
    int idx  = ((2 * kp) & 7) + ((kp >= 8) ? 8 : 0);
    bLds[j] = (unsigned)(((((c >> 4) * 32) + lane) * 16 + idx) * 2);
  }

  const char* hB  = (const char*)hbuf;
  const char* w2B = (const char*)W2;
  char* aHiB = (char*)&aBuf[0][0][0];
  char* aLoB = (char*)&aBuf[1][0][0];
  char* bHiB = (char*)&bBuf[0][0][0];
  char* bLoB = (char*)&bBuf[1][0][0];

  v8f acc[4] = {};
#pragma unroll 1
  for (int ks = 0; ks < H_DIM / 32; ++ks) {
    const unsigned kb = (unsigned)ks * 128u;
    __syncthreads();
    if (t < 128) {
#pragma unroll
      for (int j = 0; j < 8; ++j) {
        float2 f = *(const float2*)(hB + (aOff[j] + kb));
        float f0 = aVal[j] ? f.x : 0.f, f1 = aVal[j] ? f.y : 0.f;
        stagePair(aHiB + aLds[j], aLoB + aLds[j], f0, f1);
      }
    }
#pragma unroll
    for (int j = 0; j < 8; ++j) {
      float2 f = *(const float2*)(w2B + (bOff[j] + kb));
      stagePair(bHiB + bLds[j], bLoB + bLds[j], f.x, f.y);
    }
    __syncthreads();
    v16bf bh = bBuf[0][w][l];
    v16bf bl = bBuf[1][w][l];
#pragma unroll
    for (int rt = 0; rt < 4; ++rt) {
      v16bf ah = aBuf[0][rt][l];
      v16bf al = aBuf[1][rt][l];
      acc[rt] = __builtin_amdgcn_wmma_f32_16x16x32_bf16(false, ah, false, bh, (short)0, acc[rt], false, false);
      acc[rt] = __builtin_amdgcn_wmma_f32_16x16x32_bf16(false, al, false, bh, (short)0, acc[rt], false, false);
      acc[rt] = __builtin_amdgcn_wmma_f32_16x16x32_bf16(false, ah, false, bl, (short)0, acc[rt], false, false);
    }
  }

  // Epilogue: weighted scatter-add into out (exactly 2 commuting fp32 adds/token).
  const int n = l & 15;
  const int rbase = (l >> 4) * 8;
  const int m = mb + w * 16 + n;
  const float bias = b2[e * M_DIM + m];
#pragma unroll
  for (int rt = 0; rt < 4; ++rt) {
#pragma unroll
    for (int r = 0; r < 8; ++r) {
      int si = rt * 16 + rbase + r;          // slot-in-tile
      if (slot0 + si < n_e) {
        int tok = stok[si];
        float wgt = swgt[si];
        unsafeAtomicAdd(&out[(size_t)tok * M_DIM + m], wgt * (acc[rt][r] + bias));
      }
    }
  }
}

// ---------------------------------------------------------------------------
extern "C" void kernel_launch(void* const* d_in, const int* in_sizes, int n_in,
                              void* d_out, int out_size, void* d_ws, size_t ws_size,
                              hipStream_t stream) {
  const float* x  = (const float*)d_in[0];
  const float* W1 = (const float*)d_in[1];
  const float* b1 = (const float*)d_in[2];
  const float* W2 = (const float*)d_in[3];
  const float* b2 = (const float*)d_in[4];
  const float* Wr = (const float*)d_in[5];
  float* out = (float*)d_out;

  char* ws = (char*)d_ws;
  int*   counts = (int*)ws;                                        // E ints
  int*   perm   = (int*)(ws + 256);                                // E*B ints
  float* pw     = (float*)(ws + 256 + E_EXP * B_TOK * 4);          // E*B floats
  float* hbuf   = (float*)(ws + 256 + 2 * E_EXP * B_TOK * 4);      // E*B*H floats (~16.8 MB)

  hipMemsetAsync(counts, 0, E_EXP * sizeof(int), stream);
  hipMemsetAsync(out, 0, (size_t)B_TOK * M_DIM * sizeof(float), stream);

  router_kernel<<<B_TOK, 256, 0, stream>>>(x, Wr, counts, perm, pw);

  dim3 g1(B_TOK / 64, H_DIM / 128, E_EXP);   // 16 x 4 x 8
  gemm1_kernel<<<g1, 256, 0, stream>>>(x, W1, b1, counts, perm, hbuf);

  dim3 g2(B_TOK / 64, M_DIM / 80, E_EXP);    // 16 x 125 x 8
  gemm2_kernel<<<g2, 160, 0, stream>>>(hbuf, W2, b2, counts, perm, pw, out);
}
